// ReferenceModulatedCrossAttention_32212254720625
// MI455X (gfx1250) — compile-verified, matrix operand order checked by reading, not run on hardware
//
#include <hip/hip_runtime.h>
#include <hip/hip_bf16.h>

typedef __bf16 bf16_t;
typedef __attribute__((ext_vector_type(16))) __bf16 v16bf;
typedef __attribute__((ext_vector_type(8)))  __bf16 v8bf;
typedef __attribute__((ext_vector_type(4)))  __bf16 v4bf;
typedef __attribute__((ext_vector_type(8)))  float  v8f;

#define SEQ   1024
#define DIMM  1024
#define BATCH 4
#define HEADS_ 8
#define DHEAD 128
#define MROWS 4096              // BATCH*SEQ
#define ATT_SCALE 0.08838834764831845f
#define LN_EPS 1e-5f

#define USE_ASYNC 1

__device__ static inline v8f wmma_bf16(v16bf a, v16bf b, v8f c) {
  return __builtin_amdgcn_wmma_f32_16x16x32_bf16(false, a, false, b, (short)0, c, false, false);
}

// Async global->LDS 16B copy per lane (gfx1250 GLOBAL_LOAD_ASYNC_TO_LDS_B128).
__device__ static inline void async_copy_b128(bf16_t* lds_dst, const bf16_t* gsrc) {
#if USE_ASYNC
  unsigned loff = (unsigned)(unsigned long long)(uintptr_t)lds_dst; // low 32b = LDS offset
  asm volatile("global_load_async_to_lds_b128 %0, %1, off"
               :: "v"(loff), "v"(gsrc) : "memory");
#else
  *(uint4*)lds_dst = *(const uint4*)gsrc;
#endif
}
__device__ static inline void async_wait_all() {
#if USE_ASYNC
  asm volatile("s_wait_asynccnt 0x0" ::: "memory");
#endif
}

// WMMA 16x32 bf16 fragment from an LDS tile stored row-major [row][k],
// stride `ld` (elements, multiple of 8; rows 16B aligned). Works for A tiles
// ([m][k]) and for B tiles staged pre-transposed ([n][k]).
// Two ds_load_b128 per lane.
__device__ static inline v16bf load_frag16(const bf16_t* s, int ld, int r0) {
  int lane = threadIdx.x & 31;
  const bf16_t* p = s + (r0 + (lane & 15)) * ld + ((lane >> 4) * 8);
  v8bf lo = *(const v8bf*)p;
  v8bf hi = *(const v8bf*)(p + 16);
  return __builtin_shufflevector(lo, hi, 0,1,2,3,4,5,6,7,8,9,10,11,12,13,14,15);
}

// ---- LayerNorm (f32 in, bf16 out), one row of 1024 per block ---------------
__global__ __launch_bounds__(256) void ln_kernel(const float* __restrict__ x,
                                                 const float* __restrict__ w,
                                                 const float* __restrict__ b,
                                                 bf16_t* __restrict__ out) {
  int row = blockIdx.x, tid = threadIdx.x;
  const float* xr = x + (size_t)row * DIMM;
  float4 v4 = *(const float4*)(xr + tid * 4);
  float s = v4.x + v4.y + v4.z + v4.w;
  __shared__ float buf[256];
  buf[tid] = s; __syncthreads();
  for (int off = 128; off > 0; off >>= 1) { if (tid < off) buf[tid] += buf[tid+off]; __syncthreads(); }
  float mean = buf[0] * (1.0f / DIMM); __syncthreads();
  float dx = v4.x - mean, dy = v4.y - mean, dz = v4.z - mean, dw = v4.w - mean;
  buf[tid] = dx*dx + dy*dy + dz*dz + dw*dw; __syncthreads();
  for (int off = 128; off > 0; off >>= 1) { if (tid < off) buf[tid] += buf[tid+off]; __syncthreads(); }
  float rstd = rsqrtf(buf[0] * (1.0f / DIMM) + LN_EPS);
  float4 wv = *(const float4*)(w + tid * 4);
  float4 bv = *(const float4*)(b + tid * 4);
  v4bf o;
  o[0] = (bf16_t)(dx * rstd * wv.x + bv.x);
  o[1] = (bf16_t)(dy * rstd * wv.y + bv.y);
  o[2] = (bf16_t)(dz * rstd * wv.z + bv.z);
  o[3] = (bf16_t)(dw * rstd * wv.w + bv.w);
  *(v4bf*)(out + (size_t)row * DIMM + tid * 4) = o;
}

// ---- transpose + convert: in[K][N] f32 -> out[N][K] bf16 -------------------
__global__ __launch_bounds__(256) void tcvt_kernel(const float* __restrict__ in,
                                                   bf16_t* __restrict__ out,
                                                   int K, int N) {
  __shared__ float t[32][33];
  int k0 = blockIdx.y * 32, n0 = blockIdx.x * 32;
  int c = threadIdx.x & 31, r8 = threadIdx.x >> 5;
#pragma unroll
  for (int q = 0; q < 4; ++q)
    t[r8 + q*8][c] = in[(size_t)(k0 + r8 + q*8) * N + n0 + c];
  __syncthreads();
#pragma unroll
  for (int q = 0; q < 4; ++q)
    out[(size_t)(n0 + r8 + q*8) * K + k0 + c] = (bf16_t)t[c][r8 + q*8];
}

// ---- bf16 transpose per batch: in[b][R][C] -> out[b][C][R], R=C=1024 -------
__global__ __launch_bounds__(256) void tbf_kernel(const bf16_t* __restrict__ in,
                                                  bf16_t* __restrict__ out) {
  __shared__ bf16_t t[32][33];
  int b = blockIdx.z;
  const bf16_t* ib = in + (size_t)b * SEQ * DIMM;
  bf16_t* ob = out + (size_t)b * SEQ * DIMM;
  int r0 = blockIdx.y * 32, c0 = blockIdx.x * 32;
  int c = threadIdx.x & 31, r8 = threadIdx.x >> 5;
#pragma unroll
  for (int q = 0; q < 4; ++q)
    t[r8 + q*8][c] = ib[(size_t)(r0 + r8 + q*8) * DIMM + c0 + c];
  __syncthreads();
#pragma unroll
  for (int q = 0; q < 4; ++q)
    ob[(size_t)(c0 + r8 + q*8) * SEQ + r0 + c] = t[c][r8 + q*8];
}

// ---- segmented bf16 GEMM: C[M,N] = sum_s A_s[M,1024] @ B_s ------------------
// B supplied PRE-TRANSPOSED: BT[N][nseg*1024], row stride ldb.
// block tile 128x128, 8 waves (4 along M x 2 along N), wave tile 32x64.
__global__ __launch_bounds__(256) void gemm_bf16_kernel(
    const bf16_t* A0, const bf16_t* A1, const bf16_t* A2,
    const bf16_t* BT, int ldb, int nseg, int N,
    float* Cf, bf16_t* Cb, const float* bias) {
  __shared__ __align__(16) bf16_t As[128 * 40];
  __shared__ __align__(16) bf16_t Bs[128 * 40];
  int tid = threadIdx.x, lane = tid & 31, wid = tid >> 5;
  int bm = blockIdx.y * 128, bn = blockIdx.x * 128;
  int wm = (wid & 3) * 32, wn = (wid >> 2) * 64;
  v8f acc[2][4];
  v8f zero = {0.f,0.f,0.f,0.f,0.f,0.f,0.f,0.f};
#pragma unroll
  for (int i = 0; i < 2; ++i)
#pragma unroll
    for (int j = 0; j < 4; ++j) acc[i][j] = zero;

  const bf16_t* Aseg[3] = {A0, A1, A2};
  for (int s = 0; s < nseg; ++s) {
    const bf16_t* A  = Aseg[s];
    const bf16_t* Bt = BT + s * 1024;
    for (int kk = 0; kk < 1024; kk += 32) {
#pragma unroll
      for (int it = 0; it < 2; ++it) {           // 128x32 tiles, 16B per lane
        int t = tid + it * 256;
        int r = t >> 2, c = (t & 3) * 8;
        async_copy_b128(&As[r * 40 + c], A  + (size_t)(bm + r) * 1024 + kk + c);
        async_copy_b128(&Bs[r * 40 + c], Bt + (size_t)(bn + r) * ldb  + kk + c);
      }
      async_wait_all();
      __syncthreads();
      v16bf af0 = load_frag16(As, 40, wm);
      v16bf af1 = load_frag16(As, 40, wm + 16);
#pragma unroll
      for (int nf = 0; nf < 4; ++nf) {
        v16bf bfrag = load_frag16(Bs, 40, wn + nf * 16);
        acc[0][nf] = wmma_bf16(af0, bfrag, acc[0][nf]);
        acc[1][nf] = wmma_bf16(af1, bfrag, acc[1][nf]);
      }
      __syncthreads();
    }
  }
  int ln_ = lane & 15, lm = (lane >> 4) * 8;
#pragma unroll
  for (int mf = 0; mf < 2; ++mf)
#pragma unroll
    for (int nf = 0; nf < 4; ++nf)
#pragma unroll
      for (int r = 0; r < 8; ++r) {
        int m = bm + wm + mf * 16 + lm + r;
        int n = bn + wn + nf * 16 + ln_;
        float v = acc[mf][nf][r];
        if (bias) v += bias[n];
        if (Cf) Cf[(size_t)m * N + n] = v;
        else    Cb[(size_t)m * N + n] = (bf16_t)v;
      }
}

// ---- sim = (cond_bh @ refv_bh^T) * scale, per (b,h) ------------------------
__global__ __launch_bounds__(256) void sim_kernel(const bf16_t* __restrict__ condb,
                                                  const bf16_t* __restrict__ refvb,
                                                  float* __restrict__ sim) {
  __shared__ __align__(16) bf16_t As[128 * 40];
  __shared__ __align__(16) bf16_t Vs[128 * 40];
  int bh = blockIdx.z, b = bh >> 3, h = bh & 7;
  int bi = blockIdx.y * 128, bj = blockIdx.x * 128;
  int tid = threadIdx.x, lane = tid & 31, wid = tid >> 5;
  int wm = (wid & 3) * 32, wn = (wid >> 2) * 64;
  const bf16_t* A = condb + (size_t)b * SEQ * DIMM + h * DHEAD;
  const bf16_t* V = refvb + (size_t)b * SEQ * DIMM + h * DHEAD;
  v8f acc[2][4];
  v8f zero = {0.f,0.f,0.f,0.f,0.f,0.f,0.f,0.f};
#pragma unroll
  for (int i = 0; i < 2; ++i)
#pragma unroll
    for (int j = 0; j < 4; ++j) acc[i][j] = zero;

  for (int kk = 0; kk < DHEAD; kk += 32) {
#pragma unroll
    for (int it = 0; it < 2; ++it) {
      int t = tid + it * 256;
      int r = t >> 2, c = (t & 3) * 8;
      async_copy_b128(&As[r * 40 + c], A + (size_t)(bi + r) * 1024 + kk + c);
      async_copy_b128(&Vs[r * 40 + c], V + (size_t)(bj + r) * 1024 + kk + c);
    }
    async_wait_all();
    __syncthreads();
    v16bf af0 = load_frag16(As, 40, wm);
    v16bf af1 = load_frag16(As, 40, wm + 16);
#pragma unroll
    for (int nf = 0; nf < 4; ++nf) {
      // B[k][n] = refv[n][k]; refv tile is [j][k] row-major -> same loader
      v16bf bfrag = load_frag16(Vs, 40, wn + nf * 16);
      acc[0][nf] = wmma_bf16(af0, bfrag, acc[0][nf]);
      acc[1][nf] = wmma_bf16(af1, bfrag, acc[1][nf]);
    }
    __syncthreads();
  }
  int ln_ = lane & 15, lm = (lane >> 4) * 8;
  float* out = sim + (size_t)bh * SEQ * SEQ;
#pragma unroll
  for (int mf = 0; mf < 2; ++mf)
#pragma unroll
    for (int nf = 0; nf < 4; ++nf)
#pragma unroll
      for (int r = 0; r < 8; ++r) {
        int i = bi + wm + mf * 16 + lm + r;
        int j = bj + wn + nf * 16 + ln_;
        out[(size_t)i * SEQ + j] = acc[mf][nf][r] * ATT_SCALE;
      }
}

// ---- row softmax stats: per (b,h,i): max over j, 1/sum(exp) ----------------
__global__ __launch_bounds__(256) void rowstat_kernel(const float* __restrict__ sim,
                                                      float* __restrict__ stat) {
  int row = blockIdx.x, tid = threadIdx.x;
  const float* s = sim + (size_t)row * SEQ;
  float4 v4 = *(const float4*)(s + tid * 4);
  float m = fmaxf(fmaxf(v4.x, v4.y), fmaxf(v4.z, v4.w));
  __shared__ float buf[256];
  buf[tid] = m; __syncthreads();
  for (int off = 128; off > 0; off >>= 1) { if (tid < off) buf[tid] = fmaxf(buf[tid], buf[tid+off]); __syncthreads(); }
  float mx = buf[0]; __syncthreads();
  float sm = __expf(v4.x - mx) + __expf(v4.y - mx) + __expf(v4.z - mx) + __expf(v4.w - mx);
  buf[tid] = sm; __syncthreads();
  for (int off = 128; off > 0; off >>= 1) { if (tid < off) buf[tid] += buf[tid+off]; __syncthreads(); }
  if (tid == 0) { stat[(size_t)row*2] = mx; stat[(size_t)row*2+1] = 1.0f / buf[0]; }
}

// ---- column softmax stats: per (b,h,col): max over rows, 1/sum(exp) --------
__global__ __launch_bounds__(256) void colstat_kernel(const float* __restrict__ sim,
                                                      float* __restrict__ stat) {
  int bh = blockIdx.y, tid = threadIdx.x;
  int c = blockIdx.x * 32 + (tid & 31);
  int tr = tid >> 5;
  const float* s = sim + (size_t)bh * SEQ * SEQ;
  float m = -3.4e38f;
  for (int r = tr; r < SEQ; r += 8) m = fmaxf(m, s[(size_t)r * SEQ + c]);
  __shared__ float buf[256];
  __shared__ float cmax[32];
  buf[tid] = m; __syncthreads();
  if (tr == 0) {
#pragma unroll
    for (int q = 1; q < 8; ++q) m = fmaxf(m, buf[q*32 + tid]);
    cmax[tid] = m;
  }
  __syncthreads();
  float mx = cmax[tid & 31];
  float sm = 0.f;
  for (int r = tr; r < SEQ; r += 8) sm += __expf(s[(size_t)r * SEQ + c] - mx);
  __syncthreads();
  buf[tid] = sm; __syncthreads();
  if (tr == 0) {
#pragma unroll
    for (int q = 1; q < 8; ++q) sm += buf[q*32 + tid];
    stat[((size_t)bh * SEQ + c) * 2]     = mx;
    stat[((size_t)bh * SEQ + c) * 2 + 1] = 1.0f / sm;
  }
}

// ---- fused talking-heads softmax + PV --------------------------------------
// One block: batch b, 32-row i-tile, ALL 8 output heads (wave g = head g).
// VT is PRE-TRANSPOSED value matrix: VT[b][col][j] (col = h*128+d).
// transposed=0: out[i,d] = sum_j mix(softmax_row(sim))[g,i,j] * V[j, g*128+d]
// transposed=1: out[i,d] = sum_j mix(softmax_col(sim))[g,j,i] * V[j, g*128+d]
__global__ __launch_bounds__(256) void pv_kernel(
    const float* __restrict__ sim, const float* __restrict__ stat,
    const float* __restrict__ mixw, const bf16_t* __restrict__ VT,
    float* outF, bf16_t* outB, int transposed) {
  __shared__ __align__(16) bf16_t P[HEADS_][32][40]; // mixed-attn tiles per head
  __shared__ __align__(16) bf16_t Vs[1024 * 40 / 40][40]; // [col][j] 1024x32
  __shared__ float  W[64];
  __shared__ float  st[HEADS_][32][2];
  int b = blockIdx.y;
  int i0 = blockIdx.x * 32;
  int tid = threadIdx.x, lane = tid & 31, g = tid >> 5;
  if (tid < 64) W[tid] = mixw[tid];
  for (int t = tid; t < 512; t += 256) {
    int h = t >> 6, rem = t & 63, ii = rem >> 1, cc = rem & 1;
    st[h][ii][cc] = stat[(((size_t)(b * HEADS_ + h)) * SEQ + i0 + ii) * 2 + cc];
  }
  __syncthreads();
  v8f acc[2][8];
  v8f zero = {0.f,0.f,0.f,0.f,0.f,0.f,0.f,0.f};
#pragma unroll
  for (int i = 0; i < 2; ++i)
#pragma unroll
    for (int j = 0; j < 8; ++j) acc[i][j] = zero;
  const float* simb = sim + (size_t)b * HEADS_ * SEQ * SEQ;
  const bf16_t* VTb = VT + (size_t)b * SEQ * DIMM;

  for (int j0 = 0; j0 < SEQ; j0 += 32) {
    // issue async V staging: VT rows (cols of V) 0..1023, j-slice j0..j0+31
#pragma unroll
    for (int it = 0; it < 16; ++it) {
      int t = tid + it * 256;
      int r = t >> 2, c = (t & 3) * 8;
      async_copy_b128(&Vs[r][c], VTb + (size_t)r * SEQ + j0 + c);
    }
    // overlap: compute mixed-softmax tile P while DMA is in flight
    if (!transposed) {
      int il = tid >> 3, jl4 = (tid & 7) * 4;
      float p[HEADS_][4];
#pragma unroll
      for (int h = 0; h < HEADS_; ++h) {
        float4 sv = *(const float4*)&simb[((size_t)h * SEQ + (i0 + il)) * SEQ + (j0 + jl4)];
        float mx = st[h][il][0], inv = st[h][il][1];
        p[h][0] = __expf(sv.x - mx) * inv;
        p[h][1] = __expf(sv.y - mx) * inv;
        p[h][2] = __expf(sv.z - mx) * inv;
        p[h][3] = __expf(sv.w - mx) * inv;
      }
#pragma unroll
      for (int gg = 0; gg < HEADS_; ++gg) {
        float a0 = 0, a1 = 0, a2 = 0, a3 = 0;
#pragma unroll
        for (int h = 0; h < HEADS_; ++h) {
          float wgh = W[gg * 8 + h];
          a0 += wgh * p[h][0]; a1 += wgh * p[h][1];
          a2 += wgh * p[h][2]; a3 += wgh * p[h][3];
        }
        v4bf o; o[0]=(bf16_t)a0; o[1]=(bf16_t)a1; o[2]=(bf16_t)a2; o[3]=(bf16_t)a3;
        *(v4bf*)&P[gg][il][jl4] = o;
      }
    } else {
      int jl = tid >> 3, il4 = (tid & 7) * 4;
      float p[HEADS_][4];
#pragma unroll
      for (int h = 0; h < HEADS_; ++h) {
        float4 sv = *(const float4*)&simb[((size_t)h * SEQ + (j0 + jl)) * SEQ + (i0 + il4)];
        p[h][0] = __expf(sv.x - st[h][il4+0][0]) * st[h][il4+0][1];
        p[h][1] = __expf(sv.y - st[h][il4+1][0]) * st[h][il4+1][1];
        p[h][2] = __expf(sv.z - st[h][il4+2][0]) * st[h][il4+2][1];
        p[h][3] = __expf(sv.w - st[h][il4+3][0]) * st[h][il4+3][1];
      }
#pragma unroll
      for (int gg = 0; gg < HEADS_; ++gg) {
#pragma unroll
        for (int q = 0; q < 4; ++q) {
          float a = 0;
#pragma unroll
          for (int h = 0; h < HEADS_; ++h) a += W[gg * 8 + h] * p[h][q];
          P[gg][il4 + q][jl] = (bf16_t)a;
        }
      }
    }
    async_wait_all();
    __syncthreads();
    v16bf af0 = load_frag16(&P[g][0][0], 40, 0);
    v16bf af1 = load_frag16(&P[g][0][0], 40, 16);
#pragma unroll
    for (int nf = 0; nf < 8; ++nf) {
      v16bf bfrag = load_frag16(&Vs[0][0], 40, g * DHEAD + nf * 16);
      acc[0][nf] = wmma_bf16(af0, bfrag, acc[0][nf]);
      acc[1][nf] = wmma_bf16(af1, bfrag, acc[1][nf]);
    }
    __syncthreads();
  }
  int ln_ = lane & 15, lm = (lane >> 4) * 8;
#pragma unroll
  for (int mf = 0; mf < 2; ++mf)
#pragma unroll
    for (int nf = 0; nf < 8; ++nf)
#pragma unroll
      for (int r = 0; r < 8; ++r) {
        int ii = i0 + mf * 16 + lm + r;
        int col = g * DHEAD + nf * 16 + ln_;
        float v = acc[mf][nf][r];
        if (outF) outF[((size_t)b * SEQ + ii) * DIMM + col] = v;
        else      outB[((size_t)b * SEQ + ii) * DIMM + col] = (bf16_t)v;
      }
}

extern "C" void kernel_launch(void* const* d_in, const int* in_sizes, int n_in,
                              void* d_out, int out_size, void* d_ws, size_t ws_size,
                              hipStream_t stream) {
  const float* x    = (const float*)d_in[0];
  const float* cond = (const float*)d_in[1];
  const float* ref  = (const float*)d_in[2];
  const float* ln_w = (const float*)d_in[3];
  const float* ln_b = (const float*)d_in[4];
  const float* cln_w= (const float*)d_in[5];
  const float* cln_b= (const float*)d_in[6];
  const float* Wk   = (const float*)d_in[7];
  const float* Wv   = (const float*)d_in[8];
  const float* Wo   = (const float*)d_in[9];
  const float* bo   = (const float*)d_in[10];
  const float* th_w = (const float*)d_in[11];
  const float* cth_w= (const float*)d_in[12];

  char* wsp = (char*)d_ws;
  auto carve = [&](size_t bytes) { char* p = wsp; wsp += (bytes + 255) & ~(size_t)255; return p; };
  bf16_t* xn     = (bf16_t*)carve((size_t)MROWS * DIMM * 2);
  bf16_t* cn     = (bf16_t*)carve((size_t)MROWS * DIMM * 2);
  bf16_t* rn     = (bf16_t*)carve((size_t)MROWS * DIMM * 2);
  bf16_t* WkT    = (bf16_t*)carve((size_t)3072 * 1024 * 2);  // [1024][3072]
  bf16_t* WvT    = (bf16_t*)carve((size_t)2048 * 1024 * 2);  // [1024][2048]
  bf16_t* WoT    = (bf16_t*)carve((size_t)1024 * 1024 * 2);  // [1024][1024]
  bf16_t* cond_b = (bf16_t*)carve((size_t)MROWS * DIMM * 2);
  bf16_t* refv_b = (bf16_t*)carve((size_t)MROWS * DIMM * 2);
  bf16_t* cond_T = (bf16_t*)carve((size_t)MROWS * DIMM * 2); // per-b transposed
  bf16_t* refv_T = (bf16_t*)carve((size_t)MROWS * DIMM * 2);
  bf16_t* outh_b = (bf16_t*)carve((size_t)MROWS * DIMM * 2);
  float*  sim    = (float*)carve((size_t)BATCH * HEADS_ * SEQ * SEQ * 4);
  float*  rstat  = (float*)carve((size_t)BATCH * HEADS_ * SEQ * 2 * 4);
  float*  cstat  = (float*)carve((size_t)BATCH * HEADS_ * SEQ * 2 * 4);

  // LayerNorms -> bf16
  ln_kernel<<<MROWS, 256, 0, stream>>>(x,    ln_w,  ln_b,  xn);
  ln_kernel<<<MROWS, 256, 0, stream>>>(ref,  ln_w,  ln_b,  rn);
  ln_kernel<<<MROWS, 256, 0, stream>>>(cond, cln_w, cln_b, cn);
  // Weights: transpose + convert to bf16
  tcvt_kernel<<<dim3(32, 96), 256, 0, stream>>>(Wk, WkT, 3072, 1024);
  tcvt_kernel<<<dim3(32, 64), 256, 0, stream>>>(Wv, WvT, 2048, 1024);
  tcvt_kernel<<<dim3(32, 32), 256, 0, stream>>>(Wo, WoT, 1024, 1024);
  // cond = [xn|cn|rn] @ Wk  (3 K-segments)
  gemm_bf16_kernel<<<dim3(8, 32), 256, 0, stream>>>(
      xn, cn, rn, WkT, 3072, 3, 1024, nullptr, cond_b, nullptr);
  // refv = [xn|rn] @ Wv  (2 K-segments)
  gemm_bf16_kernel<<<dim3(8, 32), 256, 0, stream>>>(
      xn, rn, nullptr, WvT, 2048, 2, 1024, nullptr, refv_b, nullptr);
  // sim (scaled), per (b,h)
  sim_kernel<<<dim3(8, 8, BATCH * HEADS_), 256, 0, stream>>>(cond_b, refv_b, sim);
  // softmax statistics
  rowstat_kernel<<<BATCH * HEADS_ * SEQ, 256, 0, stream>>>(sim, rstat);
  colstat_kernel<<<dim3(32, BATCH * HEADS_), 256, 0, stream>>>(sim, cstat);
  // per-batch transposes of cond/refv for the PV B operands
  tbf_kernel<<<dim3(32, 32, BATCH), 256, 0, stream>>>(cond_b, cond_T);
  tbf_kernel<<<dim3(32, 32, BATCH), 256, 0, stream>>>(refv_b, refv_T);
  // out heads (row softmax + talking heads + @refv) -> bf16 merged
  pv_kernel<<<dim3(32, BATCH), 256, 0, stream>>>(sim, rstat, th_w, refv_T,
                                                 nullptr, outh_b, 0);
  // context_out (col softmax + talking heads + ^T @ cond) -> f32 directly
  pv_kernel<<<dim3(32, BATCH), 256, 0, stream>>>(sim, cstat, cth_w, cond_T,
                                                 (float*)d_out + (size_t)MROWS * DIMM,
                                                 nullptr, 1);
  // out = merged @ Wo + bo -> f32
  gemm_bf16_kernel<<<dim3(8, 32), 256, 0, stream>>>(
      outh_b, nullptr, nullptr, WoT, 1024, 1, 1024,
      (float*)d_out, nullptr, bo);
}
